// AttnBlock_48747878810273
// MI455X (gfx1250) — matrix operand order repaired and checked by reading, not compiled, not run on hardware
//
#include <hip/hip_runtime.h>
#include <math.h>

// ---------------- types ----------------
typedef __bf16 v16bf __attribute__((ext_vector_type(16)));
typedef __bf16 v8bf  __attribute__((ext_vector_type(8)));
typedef float  v8f   __attribute__((ext_vector_type(8)));

#define C_DIM 512
#define N_TOK 4096   // 64*64
#define BATCH 2
#define NGRP  32
#define GSIZE 16     // channels per group

// ---------------- WMMA tile loaders ----------------
// A operand: 16x32 bf16 tile from row-major [M][K] memory (K contiguous).
// Per ISA 7.12.2: lanes 0-15 (M=lane): elems 0..7 = K0..7, elems 8..15 = K16..23
//                 lanes16-31 (M=lane-16): elems 0..7 = K8..15, elems 8..15 = K24..31
__device__ __forceinline__ v16bf load_atile(const __bf16* p, int ld, int lane) {
    int r    = lane & 15;
    int koff = (lane >> 4) << 3;      // 0 or 8
    const __bf16* row = p + (size_t)r * ld + koff;
    v8bf lo = *(const v8bf*)(row);        // K koff..koff+7
    v8bf hi = *(const v8bf*)(row + 16);   // K koff+16..koff+23
    return __builtin_shufflevector(lo, hi, 0,1,2,3,4,5,6,7,8,9,10,11,12,13,14,15);
}

// B operand: 32x16 bf16 tile; memory is Bt row-major [N][K] (K contiguous).
// lanes 0-15 (N=lane) hold K=0..7,16..23, lanes 16-31 hold K=8..15,24..31.
__device__ __forceinline__ v16bf load_btile(const __bf16* p, int ld, int lane) {
    int n    = lane & 15;
    int koff = (lane >> 4) << 3;      // 0 or 8
    const __bf16* col = p + (size_t)n * ld + koff;
    v8bf lo = *(const v8bf*)(col);
    v8bf hi = *(const v8bf*)(col + 16);
    return __builtin_shufflevector(lo, hi, 0,1,2,3,4,5,6,7,8,9,10,11,12,13,14,15);
}

// ---------------- 1. GroupNorm statistics ----------------
__global__ __launch_bounds__(256) void gn_stats_kernel(const float* __restrict__ x,
                                                       float* __restrict__ mean,
                                                       float* __restrict__ rstd) {
    int bg = blockIdx.x;                       // 0..63
    const float* p = x + (size_t)bg * GSIZE * N_TOK;
    float s = 0.f, ss = 0.f;
    for (int i = threadIdx.x; i < GSIZE * N_TOK; i += 256) {
        float v = p[i]; s += v; ss += v * v;
    }
    __shared__ float rs[256], rq[256];
    rs[threadIdx.x] = s; rq[threadIdx.x] = ss;
    __syncthreads();
    for (int o = 128; o > 0; o >>= 1) {
        if (threadIdx.x < o) { rs[threadIdx.x] += rs[threadIdx.x + o]; rq[threadIdx.x] += rq[threadIdx.x + o]; }
        __syncthreads();
    }
    if (threadIdx.x == 0) {
        float m  = rs[0] * (1.0f / (GSIZE * N_TOK));
        float va = rq[0] * (1.0f / (GSIZE * N_TOK)) - m * m;
        mean[bg] = m;
        rstd[bg] = rsqrtf(va + 1e-6f);
    }
}

// ---------------- 2. GN apply + transpose -> hn_t (b,n,c) bf16 ----------------
__global__ __launch_bounds__(256) void gn_apply_kernel(const float* __restrict__ x,
                                                       const float* __restrict__ gamma,
                                                       const float* __restrict__ beta,
                                                       const float* __restrict__ mean,
                                                       const float* __restrict__ rstd,
                                                       __bf16* __restrict__ hn_t) {
    __shared__ __bf16 tile[32][33];
    int b  = blockIdx.z;
    int n0 = blockIdx.x * 32;
    int c0 = blockIdx.y * 32;
    int tx = threadIdx.x;          // 0..31
    for (int cc = threadIdx.y; cc < 32; cc += 8) {
        int c = c0 + cc;
        int g = b * NGRP + (c >> 4);
        float m = mean[g], r = rstd[g];
        float ga = gamma[c], be = beta[c];
        float v = x[((size_t)(b * C_DIM + c)) * N_TOK + n0 + tx];
        tile[cc][tx] = (__bf16)(((v - m) * r) * ga + be);
    }
    __syncthreads();
    for (int nn = threadIdx.y; nn < 32; nn += 8) {
        hn_t[((size_t)(b * N_TOK + n0 + nn)) * C_DIM + c0 + tx] = tile[tx][nn];
    }
}

// ---------------- 3. fp32 -> bf16 weight convert ----------------
__global__ __launch_bounds__(256) void cvt_bf16_kernel(const float* __restrict__ in,
                                                       __bf16* __restrict__ out, int n) {
    int i = blockIdx.x * 256 + threadIdx.x;
    if (i < n) out[i] = (__bf16)in[i];
}

// ---------------- 4. Generic WMMA GEMM (ping-pong unroll-by-2 K loop) --------
// C[M,N] = (A[M,K] x Bt[N,K]^T + bias) * alpha (+ residual).  K % 64 == 0.
__global__ __launch_bounds__(128) void wmma_gemm_kernel(
    const __bf16* __restrict__ A,  int lda, long long strideA,
    const __bf16* __restrict__ Bt, int ldb, long long strideB,
    void* __restrict__ Cout,       int ldc, long long strideC,
    int M, int N, int K,
    const float* __restrict__ bias, int bias_mode,   // 0 none, 1 per-col(N), 2 per-row(M)
    float alpha,
    const float* __restrict__ residual, long long strideR,
    int out_f32)
{
    int lane = threadIdx.x & 31;
    int wave = threadIdx.x >> 5;
    int bz   = blockIdx.z;
    const __bf16* Ab  = A  + bz * strideA;
    const __bf16* Btb = Bt + bz * strideB;
    int n0 = blockIdx.x * 64;
    int m0 = blockIdx.y * 64 + wave * 16;
    const __bf16* Aptr = Ab + (size_t)m0 * lda;

    v8f acc[4];
    #pragma unroll
    for (int nt = 0; nt < 4; ++nt)
        #pragma unroll
        for (int r = 0; r < 8; ++r) acc[nt][r] = 0.f;

    // ping buffers: k0 tiles
    v16bf a0 = load_atile(Aptr, lda, lane);
    v16bf b0[4];
    #pragma unroll
    for (int nt = 0; nt < 4; ++nt)
        b0[nt] = load_btile(Btb + (size_t)(n0 + nt * 16) * ldb, ldb, lane);
    v16bf a1;
    v16bf b1[4];

    for (int k0 = 0; k0 < K; k0 += 64) {
        int ka = k0 + 32;
        // pong loads (always in range: K % 64 == 0)
        a1 = load_atile(Aptr + ka, lda, lane);
        #pragma unroll
        for (int nt = 0; nt < 4; ++nt)
            b1[nt] = load_btile(Btb + (size_t)(n0 + nt * 16) * ldb + ka, ldb, lane);
        #pragma unroll
        for (int nt = 0; nt < 4; ++nt)
            acc[nt] = __builtin_amdgcn_wmma_f32_16x16x32_bf16(
                false, a0, false, b0[nt], (short)0, acc[nt], false, false);
        int kb = k0 + 64;
        if (kb < K) {
            __builtin_prefetch(Aptr + kb + 64, 0, 1);
            a0 = load_atile(Aptr + kb, lda, lane);
            #pragma unroll
            for (int nt = 0; nt < 4; ++nt)
                b0[nt] = load_btile(Btb + (size_t)(n0 + nt * 16) * ldb + kb, ldb, lane);
        }
        #pragma unroll
        for (int nt = 0; nt < 4; ++nt)
            acc[nt] = __builtin_amdgcn_wmma_f32_16x16x32_bf16(
                false, a1, false, b1[nt], (short)0, acc[nt], false, false);
    }

    int rowoff = (lane >> 4) << 3;
    #pragma unroll
    for (int nt = 0; nt < 4; ++nt) {
        int col  = n0 + nt * 16 + (lane & 15);
        float bc = (bias_mode == 1) ? bias[col] : 0.f;
        #pragma unroll
        for (int r = 0; r < 8; ++r) {
            int row   = m0 + r + rowoff;
            float bb  = (bias_mode == 2) ? bias[row] : bc;
            float v   = (acc[nt][r] + bb) * alpha;
            size_t idx = (size_t)row * ldc + col;
            if (out_f32) {
                float* C = (float*)Cout + bz * strideC;
                float res = residual ? residual[bz * strideR + idx] : 0.f;
                C[idx] = v + res;
            } else {
                __bf16* C = (__bf16*)Cout + bz * strideC;
                C[idx] = (__bf16)v;
            }
        }
    }
}

// ---------------- 5. Attention: 16 query rows per workgroup, S stripe in LDS ----
#define ATTN_SMEM (16 * N_TOK * 2 + 1024)
__global__ __launch_bounds__(128) void attn_kernel(const __bf16* __restrict__ q_t,
                                                   const __bf16* __restrict__ k_t,
                                                   const __bf16* __restrict__ v_c,
                                                   __bf16* __restrict__ o_t) {
    extern __shared__ char smem[];
    __bf16* sP      = (__bf16*)smem;                    // 16 x 4096 bf16 (S, then P)
    float*  sRed    = (float*)(smem + 16 * N_TOK * 2);
    float*  sWMax   = sRed;        // [4][16]
    float*  sRowM   = sRed + 64;   // [16]
    float*  sPart   = sRed + 80;   // [128]
    float*  sLinv   = sRed + 208;  // [16]

    int tid  = threadIdx.x;
    int wave = tid >> 5, lane = tid & 31;
    int b  = blockIdx.x >> 8;               // 256 row-blocks per batch
    int i0 = (blockIdx.x & 255) << 4;

    const __bf16* Q  = q_t + ((size_t)(b * N_TOK + i0)) * C_DIM;
    const __bf16* Kt = k_t + ((size_t)b * N_TOK) * C_DIM;
    const __bf16* Vc = v_c + ((size_t)b * C_DIM) * N_TOK;

    // preload the 16x512 Q stripe for this block (16 k-chunks of 32)
    v16bf qreg[16];
    #pragma unroll
    for (int kc = 0; kc < 16; ++kc)
        qreg[kc] = load_atile(Q + kc * 32, C_DIM, lane);

    // ---- stage 1: S = Q K^T for j in [wave*1024, wave*1024+1024) ----
    float rmax[8];
    #pragma unroll
    for (int r = 0; r < 8; ++r) rmax[r] = -1e30f;
    int jbase = wave * 1024;
    int rowoff = (lane >> 4) << 3;
    for (int jt = 0; jt < 64; ++jt) {
        int j0 = jbase + jt * 16;
        const __bf16* kbase = Kt + (size_t)j0 * C_DIM;
        __builtin_prefetch(kbase + 16 * C_DIM, 0, 1);
        v8f acc;
        #pragma unroll
        for (int r = 0; r < 8; ++r) acc[r] = 0.f;
        // two-deep pipeline on B tiles along the fully-unrolled c-chain
        v16bf bt0 = load_btile(kbase, C_DIM, lane);
        v16bf bt1 = load_btile(kbase + 32, C_DIM, lane);
        #pragma unroll
        for (int kc = 0; kc < 16; ++kc) {
            v16bf bt2 = bt0;
            if (kc + 2 < 16)
                bt2 = load_btile(kbase + (kc + 2) * 32, C_DIM, lane);
            acc = __builtin_amdgcn_wmma_f32_16x16x32_bf16(
                false, qreg[kc], false, bt0, (short)0, acc, false, false);
            bt0 = bt1; bt1 = bt2;
        }
        int col = j0 + (lane & 15);
        #pragma unroll
        for (int r = 0; r < 8; ++r) {
            float s = acc[r];
            rmax[r] = fmaxf(rmax[r], s);
            sP[(size_t)(r + rowoff) * N_TOK + col] = (__bf16)s;
        }
    }
    // cross-lane max within each 16-lane half (each half owns rows 0-7 / 8-15)
    #pragma unroll
    for (int m = 1; m < 16; m <<= 1)
        #pragma unroll
        for (int r = 0; r < 8; ++r)
            rmax[r] = fmaxf(rmax[r], __shfl_xor(rmax[r], m, 32));
    if ((lane & 15) == 0) {
        #pragma unroll
        for (int r = 0; r < 8; ++r) sWMax[wave * 16 + rowoff + r] = rmax[r];
    }
    __syncthreads();
    if (tid < 16) {
        float m = sWMax[tid];
        for (int w = 1; w < 4; ++w) m = fmaxf(m, sWMax[w * 16 + tid]);
        sRowM[tid] = m;
    }
    __syncthreads();

    // ---- stage 2: softmax (exp + row sums) in LDS; P overwrites S (vectorized) ----
    {
        int row = tid >> 3, chunk = tid & 7;
        float m = sRowM[row];
        float sum = 0.f;
        v8bf* rp = (v8bf*)(sP + (size_t)row * N_TOK + chunk * 512);
        for (int j = 0; j < 64; ++j) {          // 64 x 8 bf16 = 512
            v8bf v = rp[j];
            v8bf pv;
            #pragma unroll
            for (int e = 0; e < 8; ++e) {
                float p = __expf((float)v[e] - m);
                sum += p;
                pv[e] = (__bf16)p;
            }
            rp[j] = pv;
        }
        sPart[tid] = sum;
    }
    __syncthreads();
    if (tid < 16) {
        float s = 0.f;
        for (int c = 0; c < 8; ++c) s += sPart[tid * 8 + c];
        sLinv[tid] = 1.0f / s;
    }
    __syncthreads();

    // ---- stage 3: O = P V^T ; wave owns c in [wave*128, wave*128+128) ----
    // ping-pong unroll-by-2 over the 128 j-chunks (no buffer rotation moves)
    int c0 = wave * 128;
    v8f acc[8];
    #pragma unroll
    for (int nt = 0; nt < 8; ++nt)
        #pragma unroll
        for (int r = 0; r < 8; ++r) acc[nt][r] = 0.f;

    v16bf a0 = load_atile(sP, N_TOK, lane);                 // ds loads
    v16bf b0[8];
    #pragma unroll
    for (int nt = 0; nt < 8; ++nt)
        b0[nt] = load_btile(Vc + (size_t)(c0 + nt * 16) * N_TOK, N_TOK, lane);
    v16bf a1;
    v16bf b1[8];

    for (int kj = 0; kj < 128; kj += 2) {
        int ka = (kj + 1) * 32;        // always < 4096
        a1 = load_atile(sP + ka, N_TOK, lane);
        #pragma unroll
        for (int nt = 0; nt < 8; ++nt)
            b1[nt] = load_btile(Vc + (size_t)(c0 + nt * 16) * N_TOK + ka, N_TOK, lane);
        #pragma unroll
        for (int nt = 0; nt < 8; ++nt)
            acc[nt] = __builtin_amdgcn_wmma_f32_16x16x32_bf16(
                false, a0, false, b0[nt], (short)0, acc[nt], false, false);
        int kb = (kj + 2) * 32;
        if (kj + 2 < 128) {
            a0 = load_atile(sP + kb, N_TOK, lane);
            #pragma unroll
            for (int nt = 0; nt < 8; ++nt)
                b0[nt] = load_btile(Vc + (size_t)(c0 + nt * 16) * N_TOK + kb, N_TOK, lane);
        }
        #pragma unroll
        for (int nt = 0; nt < 8; ++nt)
            acc[nt] = __builtin_amdgcn_wmma_f32_16x16x32_bf16(
                false, a1, false, b1[nt], (short)0, acc[nt], false, false);
    }
    // epilogue: scale by 1/l per row, store o_t (b,i,c) bf16
    #pragma unroll
    for (int nt = 0; nt < 8; ++nt) {
        int c = c0 + nt * 16 + (lane & 15);
        #pragma unroll
        for (int r = 0; r < 8; ++r) {
            int row = r + rowoff;
            float v = acc[nt][r] * sLinv[row];
            o_t[((size_t)(b * N_TOK + i0 + row)) * C_DIM + c] = (__bf16)v;
        }
    }
}

// ---------------- host ----------------
extern "C" void kernel_launch(void* const* d_in, const int* in_sizes, int n_in,
                              void* d_out, int out_size, void* d_ws, size_t ws_size,
                              hipStream_t stream) {
    const float* x     = (const float*)d_in[0];
    const float* gam   = (const float*)d_in[1];
    const float* bet   = (const float*)d_in[2];
    const float* wq    = (const float*)d_in[3];
    const float* bq    = (const float*)d_in[4];
    const float* wk    = (const float*)d_in[5];
    const float* bk    = (const float*)d_in[6];
    const float* wv    = (const float*)d_in[7];
    const float* bv    = (const float*)d_in[8];
    const float* wo    = (const float*)d_in[9];
    const float* bo    = (const float*)d_in[10];
    float* out = (float*)d_out;

    char* ws = (char*)d_ws;
    size_t off = 0;
    auto alloc = [&](size_t bytes) -> char* {
        char* p = ws + off;
        off = (off + bytes + 255) & ~(size_t)255;
        return p;
    };
    float*  mean  = (float*)alloc(BATCH * NGRP * 4);
    float*  rstd  = (float*)alloc(BATCH * NGRP * 4);
    __bf16* wq_b  = (__bf16*)alloc((size_t)C_DIM * C_DIM * 2);
    __bf16* wk_b  = (__bf16*)alloc((size_t)C_DIM * C_DIM * 2);
    __bf16* wv_b  = (__bf16*)alloc((size_t)C_DIM * C_DIM * 2);
    __bf16* wo_b  = (__bf16*)alloc((size_t)C_DIM * C_DIM * 2);
    __bf16* hn_t  = (__bf16*)alloc((size_t)BATCH * N_TOK * C_DIM * 2);
    __bf16* q_t   = (__bf16*)alloc((size_t)BATCH * N_TOK * C_DIM * 2);
    __bf16* k_t   = (__bf16*)alloc((size_t)BATCH * N_TOK * C_DIM * 2);
    __bf16* v_c   = (__bf16*)alloc((size_t)BATCH * C_DIM * N_TOK * 2);
    __bf16* o_t   = (__bf16*)alloc((size_t)BATCH * N_TOK * C_DIM * 2);
    (void)ws_size; (void)n_in; (void)in_sizes; (void)out_size;

    // 1. GroupNorm stats
    gn_stats_kernel<<<BATCH * NGRP, 256, 0, stream>>>(x, mean, rstd);
    // 2. GN apply + transpose
    gn_apply_kernel<<<dim3(N_TOK / 32, C_DIM / 32, BATCH), dim3(32, 8), 0, stream>>>(
        x, gam, bet, mean, rstd, hn_t);
    // 3. weight conversion
    int wn = C_DIM * C_DIM;
    cvt_bf16_kernel<<<(wn + 255) / 256, 256, 0, stream>>>(wq, wq_b, wn);
    cvt_bf16_kernel<<<(wn + 255) / 256, 256, 0, stream>>>(wk, wk_b, wn);
    cvt_bf16_kernel<<<(wn + 255) / 256, 256, 0, stream>>>(wv, wv_b, wn);
    cvt_bf16_kernel<<<(wn + 255) / 256, 256, 0, stream>>>(wo, wo_b, wn);

    const float scale = 0.044194173824159216f;  // 512^-0.5 folded into Q

    // 4a. Q = hn_t x wq^T  (M=8192,N=512,K=512), q' = (q + bq)*scale
    wmma_gemm_kernel<<<dim3(C_DIM / 64, (BATCH * N_TOK) / 64, 1), 128, 0, stream>>>(
        hn_t, C_DIM, 0, wq_b, C_DIM, 0, q_t, C_DIM, 0,
        BATCH * N_TOK, C_DIM, C_DIM, bq, 1, scale, nullptr, 0, 0);
    // 4b. K
    wmma_gemm_kernel<<<dim3(C_DIM / 64, (BATCH * N_TOK) / 64, 1), 128, 0, stream>>>(
        hn_t, C_DIM, 0, wk_b, C_DIM, 0, k_t, C_DIM, 0,
        BATCH * N_TOK, C_DIM, C_DIM, bk, 1, 1.0f, nullptr, 0, 0);
    // 4c. V in (b,c,n) layout: A=wv, Bt=hn_t per batch
    wmma_gemm_kernel<<<dim3(N_TOK / 64, C_DIM / 64, BATCH), 128, 0, stream>>>(
        wv_b, C_DIM, 0, hn_t, C_DIM, (long long)N_TOK * C_DIM,
        v_c, N_TOK, (long long)C_DIM * N_TOK,
        C_DIM, N_TOK, C_DIM, bv, 2, 1.0f, nullptr, 0, 0);

    // 5. attention (16 rows per workgroup; 128 KB score stripe in LDS,
    //    2 workgroups per 320 KB WGP)
    hipFuncSetAttribute((const void*)attn_kernel,
                        hipFuncAttributeMaxDynamicSharedMemorySize, ATTN_SMEM);
    attn_kernel<<<BATCH * (N_TOK / 16), 128, ATTN_SMEM, stream>>>(q_t, k_t, v_c, o_t);

    // 6. out = wo x O^T + bo + x  (fp32)
    wmma_gemm_kernel<<<dim3(N_TOK / 64, C_DIM / 64, BATCH), 128, 0, stream>>>(
        wo_b, C_DIM, 0, o_t, C_DIM, (long long)N_TOK * C_DIM,
        out, N_TOK, (long long)C_DIM * N_TOK,
        C_DIM, N_TOK, C_DIM, bo, 2, 1.0f, x, (long long)C_DIM * N_TOK, 1);
}
